// Quantizer_55989193670843
// MI455X (gfx1250) — compile-verified, matrix operand order checked by reading, not run on hardware
//
#include <hip/hip_runtime.h>

typedef __attribute__((ext_vector_type(16))) __bf16 v16bf;
typedef __attribute__((ext_vector_type(8)))  __bf16 v8bf;
typedef __attribute__((ext_vector_type(8)))  float  v8f;

static constexpr int NQ   = 8;
static constexpr int NG   = 2;
static constexpr int NC   = 1024;
static constexpr int ED   = 256;
static constexpr int BB   = 16;
static constexpr int TT   = 2048;
static constexpr int DD   = NG * ED;        // 512
static constexpr int NTOK = BB * TT;        // 32768

union V16U { v16bf v; v8bf h[2]; };

static __device__ __forceinline__ unsigned short f2bf(float f) {
    unsigned u = __float_as_uint(f);
    unsigned r = u + 0x7fffu + ((u >> 16) & 1u);   // round-to-nearest-even
    return (unsigned short)(r >> 16);
}

// ---------------- init: zero quantized output + loss accumulators ----------
__global__ void k_zero(float4* __restrict__ q, float* __restrict__ acc) {
    size_t i = (size_t)blockIdx.x * blockDim.x + threadIdx.x;
    q[i] = make_float4(0.f, 0.f, 0.f, 0.f);
    if (blockIdx.x == 0 && threadIdx.x < 16) acc[threadIdx.x] = 0.f;
}

// ---------------- emb f32 -> bf16 + per-code squared norms -----------------
__global__ void k_convert(const float* __restrict__ emb,
                          unsigned short* __restrict__ embb,
                          float* __restrict__ norms) {
    int row = blockIdx.x;                 // NQ*NG*NC rows of ED
    int lane = threadIdx.x;               // 32 threads (1 wave)
    size_t base = (size_t)row * ED;
    float s = 0.f;
#pragma unroll
    for (int j = 0; j < 8; ++j) {
        int i = lane + j * 32;
        float v = emb[base + i];
        embb[base + i] = f2bf(v);
        s += v * v;
    }
#pragma unroll
    for (int m = 16; m >= 1; m >>= 1) s += __shfl_xor(s, m, 32);
    if (lane == 0) norms[row] = s;
}

// ---------------- layer-0 codebook cross-entropy ---------------------------
__global__ void k_eloss(const float* __restrict__ emb0,
                        const float* __restrict__ norms,
                        float* __restrict__ eacc) {
    int g = blockIdx.x >> 10;
    int c = blockIdx.x & 1023;
    const float* Eg = emb0 + (size_t)g * NC * ED;
    __shared__ float ec[ED];
    __shared__ float sred[8];
    int tid = threadIdx.x, lane = tid & 31, wave = tid >> 5;
    ec[tid] = Eg[(size_t)c * ED + tid];
    __syncthreads();

    float lg[4];
    float mx = -3.4e38f;
#pragma unroll
    for (int j = 0; j < 4; ++j) {
        int d = tid + j * 256;
        const float* Ed = Eg + (size_t)d * ED;
        float dot = 0.f;
        for (int e = 0; e < ED; ++e) dot += ec[e] * Ed[e];
        lg[j] = 3.0f * dot;
        mx = fmaxf(mx, lg[j]);
    }
#pragma unroll
    for (int m = 16; m >= 1; m >>= 1) mx = fmaxf(mx, __shfl_xor(mx, m, 32));
    if (lane == 0) sred[wave] = mx;
    __syncthreads();
    float bm = sred[0];
#pragma unroll
    for (int w = 1; w < 8; ++w) bm = fmaxf(bm, sred[w]);
    __syncthreads();
    float se = 0.f;
#pragma unroll
    for (int j = 0; j < 4; ++j) se += expf(lg[j] - bm);
#pragma unroll
    for (int m = 16; m >= 1; m >>= 1) se += __shfl_xor(se, m, 32);
    if (lane == 0) sred[wave] = se;
    __syncthreads();
    if (tid == 0) {
        float s = 0.f;
        for (int w = 0; w < 8; ++w) s += sred[w];
        float lse  = bm + logf(s);
        float diag = 3.0f * norms[(size_t)g * NC + c];
        atomicAdd(eacc, lse - diag);
    }
}

// Issue one 16-code E tile (8 KB) as async global->LDS copies (ASYNCcnt).
static __device__ __forceinline__ void prefetch_tile(unsigned long long srcBase,
                                                     unsigned esDst, int tid, int c0) {
#pragma unroll
    for (int r = 0; r < 4; ++r) {
        int chunk = r * 128 + tid;            // 512 x 16B chunks per tile
        int code = chunk >> 5;
        int off  = (chunk & 31) * 8;          // bf16 units
        unsigned voff = (unsigned)(((c0 + code) * ED + off) * 2);     // global byte off
        unsigned lds  = esDst + (unsigned)((code * 264 + off) * 2);   // LDS byte addr
        asm volatile("global_load_async_to_lds_b128 %0, %1, %2 offset:0"
                     :: "v"(lds), "v"(voff), "s"(srcBase) : "memory");
    }
}

// ---------------- main per-layer kernel: WMMA distances + argmin + update --
__global__ __launch_bounds__(128)
void k_layer(const float* __restrict__ embf, const unsigned short* __restrict__ embb,
             const float* __restrict__ norms, float* __restrict__ residual,
             float* __restrict__ quant, float* __restrict__ idxOut,
             float* __restrict__ acc, int layer) {
    int tile = blockIdx.x;          // 512 tiles of 64 tokens
    int g    = blockIdx.y;          // group
    int b    = tile >> 5;
    int t0   = (tile & 31) << 6;
    int tid  = threadIdx.x, lane = tid & 31, wave = tid >> 5;
    int l15  = lane & 15, hl = lane >> 4;

    __shared__ __align__(16) unsigned short Xs[64][264];      // 64 tokens x 256 dims
    __shared__ __align__(16) unsigned short Es[2][16][264];   // double-buffered E tiles
    __shared__ int   idxS[64];
    __shared__ float sred[4];

    // ISA: generic LDS address carries the LDS byte offset in its low 32 bits
    unsigned esBase = (unsigned)(reinterpret_cast<unsigned long long>(&Es[0][0][0]));
    const unsigned esStride = 16u * 264u * 2u;

    unsigned long long egbAddr =
        reinterpret_cast<unsigned long long>(embb) +
        (unsigned long long)(layer * NG + g) * NC * ED * 2ull;

    // kick off tile 0 prefetch before staging X (overlaps with X loads)
    prefetch_tile(egbAddr, esBase, tid, 0);

    size_t rowBase = ((size_t)(b * DD + g * ED)) * TT + t0;   // + e*TT + tok

    // stage residual tile -> LDS bf16 (coalesced 64-wide along t)
    for (int rep = 0; rep < 128; ++rep) {
        int flat = rep * 128 + tid;
        int e = flat >> 6, tok = flat & 63;
        float v = residual[rowBase + (size_t)e * TT + tok];
        Xs[tok][e] = f2bf(v);
    }
    __syncthreads();

    // A-fragments (ISA 16-bit A 16x32 layout): lane row M=l15, K per-half pattern
    V16U a[8];
    int tokl = wave * 16 + l15;
#pragma unroll
    for (int kc = 0; kc < 8; ++kc) {
        int d0 = kc * 32 + hl * 8;
        a[kc].h[0] = *(const v8bf*)&Xs[tokl][d0];
        a[kc].h[1] = *(const v8bf*)&Xs[tokl][d0 + 16];
    }

    float rmin[8]; int ridx[8];
#pragma unroll
    for (int v = 0; v < 8; ++v) { rmin[v] = 3.4e38f; ridx[v] = 0; }

    const float* nrmB = norms + (size_t)(layer * NG + g) * NC;

    for (int cb = 0; cb < 64; ++cb) {
        int c0  = cb * 16;
        int cur = cb & 1;
        // my async copies for the current buffer have landed in LDS...
        asm volatile("s_wait_asynccnt 0x0" ::: "memory");
        // ...and after the barrier, everyone's have
        __syncthreads();
        if (cb + 1 < 64)
            prefetch_tile(egbAddr, esBase + (unsigned)((cb + 1) & 1) * esStride,
                          tid, (cb + 1) * 16);

        float nrm = nrmB[c0 + l15];
        v8f accv = {0.f, 0.f, 0.f, 0.f, 0.f, 0.f, 0.f, 0.f};
#pragma unroll
        for (int kc = 0; kc < 8; ++kc) {
            int d0 = kc * 32 + hl * 8;
            V16U bb;
            bb.h[0] = *(const v8bf*)&Es[cur][l15][d0];
            bb.h[1] = *(const v8bf*)&Es[cur][l15][d0 + 16];
            accv = __builtin_amdgcn_wmma_f32_16x16x32_bf16(
                false, a[kc].v, false, bb.v, (short)0, accv, false, false);
        }
#pragma unroll
        for (int v = 0; v < 8; ++v) {
            float val = nrm - 2.0f * accv[v];   // ||E||^2 - 2 x.E
            if (val < rmin[v]) { rmin[v] = val; ridx[v] = c0 + l15; }
        }
    }

    // argmin across the 16 code-lanes of each half (token rows stay in-half)
#pragma unroll
    for (int v = 0; v < 8; ++v) {
        float val = rmin[v]; int id = ridx[v];
#pragma unroll
        for (int m = 1; m < 16; m <<= 1) {
            float ov = __shfl_xor(val, m, 32);
            int  oid = __shfl_xor(id, m, 32);
            if (ov < val || (ov == val && oid < id)) { val = ov; id = oid; }
        }
        rmin[v] = val; ridx[v] = id;
    }
    if (l15 < 8) {
        int sel = 0;
#pragma unroll
        for (int v = 0; v < 8; ++v) if (l15 == v) sel = ridx[v];
        int mrow = l15 + 8 * hl;             // token row in C/D layout
        int tokLoc = wave * 16 + mrow;
        idxS[tokLoc] = sel;
        size_t n = (size_t)b * TT + t0 + tokLoc;
        idxOut[(size_t)g * NTOK + n] = (float)sel;
    }
    __syncthreads();

    // fused straight-through update + loss accumulation
    const float* Ef = embf + (size_t)(layer * NG + g) * NC * ED;
    float ssq = 0.f;
    for (int rep = 0; rep < 128; ++rep) {
        int flat = rep * 128 + tid;
        int e = flat >> 6, tok = flat & 63;
        int code = idxS[tok];
        float z = Ef[(size_t)code * ED + e];
        size_t off = rowBase + (size_t)e * TT + tok;
        float rr = residual[off] - z;
        residual[off] = rr;
        quant[off] += z;
        ssq += rr * rr;
    }
#pragma unroll
    for (int m = 16; m >= 1; m >>= 1) ssq += __shfl_xor(ssq, m, 32);
    if (lane == 0) sred[wave] = ssq;
    __syncthreads();
    if (tid == 0) atomicAdd(acc + layer, sred[0] + sred[1] + sred[2] + sred[3]);
}

// ---------------- final scalar loss ----------------------------------------
__global__ void k_final(const float* __restrict__ acc, float* __restrict__ outLoss) {
    if (threadIdx.x == 0 && blockIdx.x == 0) {
        float s = 0.f;
        for (int i = 0; i < NQ; ++i) {
            float li = 1.25f * acc[i] / (float)((size_t)BB * DD * TT);
            if (i == 0) li += 0.1f * acc[8] / (float)(NG * NC);
            s += li;
        }
        *outLoss = s * (1.0f / (float)NQ);
    }
}

extern "C" void kernel_launch(void* const* d_in, const int* in_sizes, int n_in,
                              void* d_out, int out_size, void* d_ws, size_t ws_size,
                              hipStream_t stream) {
    const float* xin = (const float*)d_in[0];
    const float* emb = (const float*)d_in[1];
    float* out = (float*)d_out;

    char* ws = (char*)d_ws;
    float*          residual = (float*)ws;                                   // 64 MiB
    unsigned short* embb     = (unsigned short*)(ws + 67108864);             // 8 MiB
    float*          norms    = (float*)(ws + 67108864 + 8388608);            // 64 KiB
    float*          acc      = (float*)(ws + 67108864 + 8388608 + 65536);    // 16 f32

    const size_t QN = (size_t)BB * DD * TT;          // 16,777,216

    hipMemcpyAsync(residual, xin, QN * sizeof(float), hipMemcpyDeviceToDevice, stream);
    k_zero<<<16384, 256, 0, stream>>>((float4*)out, acc);
    k_convert<<<NQ * NG * NC, 32, 0, stream>>>(emb, embb, norms);
    k_eloss<<<NG * NC, 256, 0, stream>>>(emb, norms, acc + 8);
    for (int i = 0; i < NQ; ++i) {
        float* idxBase = out + QN + 1 + (size_t)i * NG * NTOK;
        k_layer<<<dim3(512, 2), 128, 0, stream>>>(emb, embb, norms, residual,
                                                  out, idxBase, acc, i);
    }
    k_final<<<1, 32, 0, stream>>>(acc, out + QN);
}
